// SlotAttentionModuleV2_50036368999076
// MI455X (gfx1250) — matrix operand order repaired and checked by reading, not compiled, not run on hardware
//
#include <hip/hip_runtime.h>
#include <hip/hip_bf16.h>
#include <math.h>

#define BATCH 64
#define SEQ   4096
#define FDIM  256
#define DDIM  128
#define NSLOT 8
#define KPAD  16
#define NSEG  8
#define LN_EPS   1e-5f
#define EPS_ATTN 1e-8f
#define INV_SCALE 0.08838834764831843f  /* 1/sqrt(128) */

typedef __bf16 bf16;
typedef __attribute__((ext_vector_type(16))) __bf16 v16bf;
typedef __attribute__((ext_vector_type(8)))  float  v8f;

struct __attribute__((aligned(16))) bf16x8 { bf16 v[8]; };

__device__ inline v8f wmma_bf16_f32(v16bf a, v16bf b, v8f c) {
  return __builtin_amdgcn_wmma_f32_16x16x32_bf16(false, a, false, b, (short)0, c,
                                                 false, false);
}

// A fragment (16x32 bf16) from row-major source, row stride ld (elements).
// Requires: (m*ld), kbase multiples of 8 elements -> 16B-aligned vector loads.
// lane<16: row m=l, K kbase+0..15; lane>=16: K kbase+16..31
__device__ inline v16bf load_a_frag(const bf16* src, int ld, int kbase) {
  const int l = threadIdx.x & 31;
  const int m = l & 15;
  const int koff = (l < 16) ? 0 : 16;
  const bf16* p = src + (size_t)m * ld + kbase + koff;
  bf16x8 lo = *reinterpret_cast<const bf16x8*>(p);
  bf16x8 hi = *reinterpret_cast<const bf16x8*>(p + 8);
  v16bf a;
#pragma unroll
  for (int j = 0; j < 8; ++j) { a[j] = lo.v[j]; a[8 + j] = hi.v[j]; }
  return a;
}

// B fragment (32x16) where B is stored COLUMN-major (i.e. source is N x K
// row-major with row stride ld): contiguous 16B loads per half.
// lane<16: col n=l, rows kbase+0..15; lane>=16: rows kbase+16..31
__device__ inline v16bf load_b_colmajor(const bf16* srcT, int ld, int kbase) {
  const int l = threadIdx.x & 31;
  const int n = l & 15;
  const int koff = (l < 16) ? 0 : 16;
  const bf16* p = srcT + (size_t)n * ld + kbase + koff;
  bf16x8 lo = *reinterpret_cast<const bf16x8*>(p);
  bf16x8 hi = *reinterpret_cast<const bf16x8*>(p + 8);
  v16bf b;
#pragma unroll
  for (int j = 0; j < 8; ++j) { b[j] = lo.v[j]; b[8 + j] = hi.v[j]; }
  return b;
}

// Same but columns >= nvalid are zero (for q padded to 16 slots).
__device__ inline v16bf load_b_colmajor_guard(const bf16* srcT, int ld, int kbase,
                                              int nvalid) {
  const int l = threadIdx.x & 31;
  const int n = l & 15;
  const int koff = (l < 16) ? 0 : 16;
  v16bf b;
  if (n < nvalid) {
    const bf16* p = srcT + (size_t)n * ld + kbase + koff;
    bf16x8 lo = *reinterpret_cast<const bf16x8*>(p);
    bf16x8 hi = *reinterpret_cast<const bf16x8*>(p + 8);
#pragma unroll
    for (int j = 0; j < 8; ++j) { b[j] = lo.v[j]; b[8 + j] = hi.v[j]; }
  } else {
#pragma unroll
    for (int j = 0; j < 16; ++j) b[j] = (bf16)0.0f;
  }
  return b;
}

// Pre-permuted weight fragment: one contiguous 32B load per lane.
__device__ inline v16bf load_b_perm(const bf16* perm, int tile, int ch) {
  const int lane = threadIdx.x & 31;
  const v16bf* p = reinterpret_cast<const v16bf*>(
      perm + ((((size_t)tile * 8 + ch) * 32 + lane) << 4));
  return *p;
}

__device__ inline float wred_add32(float v) {
#pragma unroll
  for (int m = 16; m >= 1; m >>= 1) v += __shfl_xor(v, m, 32);
  return v;
}
__device__ inline float hred_add16(float v) {
#pragma unroll
  for (int m = 8; m >= 1; m >>= 1) v += __shfl_xor(v, m, 32);
  return v;
}
__device__ inline float hred_max16(float v) {
#pragma unroll
  for (int m = 8; m >= 1; m >>= 1) v = fmaxf(v, __shfl_xor(v, m, 32));
  return v;
}

// ---------------- prep: weights -> bf16 WMMA-B-fragment layout ----------------
// perm[((tile*8 + ch)*32 + lane)*16 + j] = W[(ch*32 + koff(lane) + j)*128 + tile*16 + (lane&15)]
__global__ void k_permute_weights(const float* __restrict__ Wk,
                                  const float* __restrict__ Wv,
                                  bf16* __restrict__ wkp, bf16* __restrict__ wvp) {
  int i = blockIdx.x * blockDim.x + threadIdx.x;
  if (i >= FDIM * DDIM) return;
  int j    = i & 15;
  int l    = (i >> 4) & 31;
  int ch   = (i >> 9) & 7;
  int tile = i >> 12;
  int n = l & 15;
  int koff = (l < 16) ? 0 : 16;
  int src = (ch * 32 + koff + j) * DDIM + tile * 16 + n;
  wkp[i] = (bf16)Wk[src];
  wvp[i] = (bf16)Wv[src];
}

__global__ void k_init_slots(const float* __restrict__ slot_init,
                             float* __restrict__ slots) {
  int i = blockIdx.x * blockDim.x + threadIdx.x;
  const int per = NSLOT * DDIM;
  if (i < BATCH * per) slots[i] = slot_init[i % per];
}

// ---------------- fused input LN + k/v projection ----------------
// k stored row-major [b][n][d]; v stored transposed [b][d][n].
#define XS_LD (FDIM + 8)   /* 264: multiple of 8 -> rows stay 16B aligned */
__global__ __launch_bounds__(256) void k_ln_kv(
    const float* __restrict__ x, const float* __restrict__ lnw,
    const float* __restrict__ lnb,
    const bf16* __restrict__ wkp, const bf16* __restrict__ wvp,
    bf16* __restrict__ kout, bf16* __restrict__ vT) {
  __shared__ bf16 xs[16 * XS_LD];
  const int wave = threadIdx.x >> 5;
  const int lane = threadIdx.x & 31;
  const size_t row0 = (size_t)blockIdx.x * 16;
  const int c0 = lane * 8;
  float w8[8], b8[8];
#pragma unroll
  for (int j = 0; j < 8; ++j) { w8[j] = lnw[c0 + j]; b8[j] = lnb[c0 + j]; }
#pragma unroll
  for (int rr = 0; rr < 2; ++rr) {
    const int rl = wave * 2 + rr;
    const float* xr = x + (row0 + rl) * FDIM + c0;
    float4 xa = *reinterpret_cast<const float4*>(xr);
    float4 xb = *reinterpret_cast<const float4*>(xr + 4);
    float xv[8] = {xa.x, xa.y, xa.z, xa.w, xb.x, xb.y, xb.z, xb.w};
    float s = 0.f, sq = 0.f;
#pragma unroll
    for (int j = 0; j < 8; ++j) { s += xv[j]; sq += xv[j] * xv[j]; }
    s = wred_add32(s); sq = wred_add32(sq);
    float mu = s * (1.0f / FDIM);
    float var = sq * (1.0f / FDIM) - mu * mu;
    float rstd = rsqrtf(var + LN_EPS);
#pragma unroll
    for (int j = 0; j < 8; ++j)
      xs[rl * XS_LD + c0 + j] = (bf16)((xv[j] - mu) * rstd * w8[j] + b8[j]);
  }
  __syncthreads();
  v8f ak = {}; v8f av = {};
#pragma unroll
  for (int ch = 0; ch < 8; ++ch) {
    v16bf a  = load_a_frag(xs, XS_LD, ch * 32);
    v16bf bk = load_b_perm(wkp, wave, ch);
    v16bf bv = load_b_perm(wvp, wave, ch);
    ak = wmma_bf16_f32(a, bk, ak);
    av = wmma_bf16_f32(a, bv, av);
  }
  const int col = wave * 16 + (lane & 15);
  const int mh = (lane < 16) ? 0 : 8;
  // k row-major (scattered b16 stores)
#pragma unroll
  for (int i = 0; i < 8; ++i)
    kout[(row0 + mh + i) * DDIM + col] = (bf16)ak[i];
  // v transposed: lane holds 8 consecutive n for fixed d -> one 16B store
  const size_t b = row0 >> 12;             // row0 / SEQ
  const size_t nbase = (row0 & (SEQ - 1)) + mh;
  bf16x8 vpack;
#pragma unroll
  for (int i = 0; i < 8; ++i) vpack.v[i] = (bf16)av[i];
  *reinterpret_cast<bf16x8*>(vT + (b * DDIM + col) * SEQ + nbase) = vpack;
}

// ---------------- per-iteration: q projection (+ zero scratch) ----------------
__global__ __launch_bounds__(128) void k_qproj(
    const float* __restrict__ slots, const float* __restrict__ lnw,
    const float* __restrict__ lnb, const float* __restrict__ Wq,
    bf16* __restrict__ qb, float* __restrict__ colsum,
    float* __restrict__ updates) {
  __shared__ float sn[NSLOT * DDIM];
  const int b = blockIdx.x;
  const int t = threadIdx.x;
  if (t < KPAD) colsum[b * KPAD + t] = 0.f;
#pragma unroll
  for (int i = 0; i < 8; ++i)
    updates[(size_t)b * NSLOT * DDIM + i * DDIM + t] = 0.f;
  const int s = t >> 4;
  const int sub = t & 15;
  const float* sr = slots + ((size_t)b * NSLOT + s) * DDIM;
  float xv[8]; float sm = 0.f, sq = 0.f;
#pragma unroll
  for (int j = 0; j < 8; ++j) { float v = sr[sub * 8 + j]; xv[j] = v; sm += v; sq += v * v; }
  sm = hred_add16(sm); sq = hred_add16(sq);
  float mu = sm * (1.0f / DDIM);
  float rstd = rsqrtf(sq * (1.0f / DDIM) - mu * mu + LN_EPS);
#pragma unroll
  for (int j = 0; j < 8; ++j) {
    int c = sub * 8 + j;
    sn[s * DDIM + c] = (xv[j] - mu) * rstd * lnw[c] + lnb[c];
  }
  __syncthreads();
  float acc[8];
#pragma unroll
  for (int i = 0; i < 8; ++i) acc[i] = 0.f;
  for (int c = 0; c < DDIM; ++c) {
    float w = Wq[c * DDIM + t];
#pragma unroll
    for (int i = 0; i < 8; ++i) acc[i] += sn[i * DDIM + c] * w;
  }
#pragma unroll
  for (int i = 0; i < 8; ++i)
    qb[((size_t)b * NSLOT + i) * DDIM + t] = (bf16)acc[i];
}

// ---------------- per-iteration: logits + softmax + attn^T store ----------------
__global__ __launch_bounds__(256) void k_attn(
    const bf16* __restrict__ kb, const bf16* __restrict__ qb,
    bf16* __restrict__ attnT, float* __restrict__ colsum) {
  __shared__ float cs[KPAD];
  const int b = blockIdx.y;
  const int wave = threadIdx.x >> 5;
  const int lane = threadIdx.x & 31;
  if (threadIdx.x < KPAD) cs[threadIdx.x] = 0.f;
  __syncthreads();
  const int n0 = blockIdx.x * 128 + wave * 16;
  const bf16* ksrc = kb + ((size_t)b * SEQ + n0) * DDIM;
  const bf16* qsrc = qb + (size_t)b * NSLOT * DDIM;
  v8f acc = {};
#pragma unroll
  for (int ch = 0; ch < 4; ++ch) {
    v16bf a  = load_a_frag(ksrc, DDIM, ch * 32);
    v16bf bq = load_b_colmajor_guard(qsrc, DDIM, ch * 32, NSLOT);
    acc = wmma_bf16_f32(a, bq, acc);
  }
  const int slot = lane & 15;
  const int half8 = (lane < 16) ? 0 : 8;
  const bool valid = (slot < NSLOT);
  bf16x8 outv;
  float lsum = 0.f;
#pragma unroll
  for (int i = 0; i < 8; ++i) {
    float lg = valid ? acc[i] * INV_SCALE : -INFINITY;
    float mx = hred_max16(lg);
    float e = __expf(lg - mx);
    float ssum = hred_add16(e);
    float a = valid ? (e / ssum + EPS_ATTN) : 0.f;
    outv.v[i] = (bf16)a;
    lsum += a;
  }
  *reinterpret_cast<bf16x8*>(attnT + ((size_t)b * KPAD + slot) * SEQ + n0 + half8) = outv;
  atomicAdd(&cs[slot], lsum);
  __syncthreads();
  if (threadIdx.x < KPAD)
    atomicAdd(&colsum[b * KPAD + threadIdx.x], cs[threadIdx.x]);
}

// ---------------- per-iteration: updates = attn^T @ v ----------------
__global__ __launch_bounds__(256) void k_updates(
    const bf16* __restrict__ attnT, const bf16* __restrict__ vT,
    float* __restrict__ updates) {
  const int b = blockIdx.y;
  const int seg = blockIdx.x;
  const int wave = threadIdx.x >> 5;
  const int lane = threadIdx.x & 31;
  const int d0 = wave * 16;
  const bf16* asrc = attnT + (size_t)b * KPAD * SEQ;
  const bf16* vsrc = vT + ((size_t)b * DDIM + d0) * SEQ;  // column-major B
  v8f acc = {};
  const int chunks = SEQ / NSEG / 32;  // 16
  const int c0 = seg * chunks;
  for (int cc = 0; cc < chunks; ++cc) {
    int nbase = (c0 + cc) * 32;
    v16bf a  = load_a_frag(asrc, SEQ, nbase);
    v16bf bv = load_b_colmajor(vsrc, SEQ, nbase);
    acc = wmma_bf16_f32(a, bv, acc);
  }
  if (lane < 16) {
    int d = d0 + lane;
#pragma unroll
    for (int i = 0; i < 8; ++i)
      atomicAdd(&updates[((size_t)b * NSLOT + i) * DDIM + d], acc[i]);
  }
}

// ---------------- per-iteration: GRU + MLP (small, scalar f32) ----------------
__global__ __launch_bounds__(128) void k_gru_mlp(
    const float* __restrict__ updates, const float* __restrict__ colsum,
    const float* __restrict__ gru_wi, const float* __restrict__ gru_wh,
    const float* __restrict__ gru_bi, const float* __restrict__ gru_bh,
    const float* __restrict__ ln_mlp_w, const float* __restrict__ ln_mlp_b,
    const float* __restrict__ mlp_w1, const float* __restrict__ mlp_b1,
    const float* __restrict__ mlp_w2, const float* __restrict__ mlp_b2,
    float* __restrict__ slots) {
  __shared__ float up[NSLOT * DDIM];
  __shared__ float sp[NSLOT * DDIM];
  __shared__ float snew[NSLOT * DDIM];
  __shared__ float nrm[NSLOT * DDIM];
  __shared__ float hbuf[NSLOT * 2 * DDIM];
  const int b = blockIdx.x;
  const int t = threadIdx.x;
#pragma unroll
  for (int i = 0; i < 8; ++i) {
    int idx = i * DDIM + t;
    up[idx] = updates[(size_t)b * NSLOT * DDIM + idx] / colsum[b * KPAD + i];
    sp[idx] = slots[(size_t)b * NSLOT * DDIM + idx];
  }
  __syncthreads();
  float gxr[8], gxz[8], gxn[8], ghr[8], ghz[8], ghn[8];
#pragma unroll
  for (int s = 0; s < 8; ++s) { gxr[s] = gxz[s] = gxn[s] = ghr[s] = ghz[s] = ghn[s] = 0.f; }
  for (int c = 0; c < DDIM; ++c) {
    const float* wi = gru_wi + (size_t)c * 3 * DDIM;
    const float* wh = gru_wh + (size_t)c * 3 * DDIM;
    float wir = wi[t], wiz = wi[t + DDIM], win = wi[t + 2 * DDIM];
    float whr = wh[t], whz = wh[t + DDIM], whn = wh[t + 2 * DDIM];
#pragma unroll
    for (int s = 0; s < 8; ++s) {
      float u = up[s * DDIM + c];
      float p = sp[s * DDIM + c];
      gxr[s] += u * wir; gxz[s] += u * wiz; gxn[s] += u * win;
      ghr[s] += p * whr; ghz[s] += p * whz; ghn[s] += p * whn;
    }
  }
  float bir = gru_bi[t], biz = gru_bi[t + DDIM], bin = gru_bi[t + 2 * DDIM];
  float bhr = gru_bh[t], bhz = gru_bh[t + DDIM], bhn = gru_bh[t + 2 * DDIM];
#pragma unroll
  for (int s = 0; s < 8; ++s) {
    float r  = 1.f / (1.f + __expf(-(gxr[s] + bir + ghr[s] + bhr)));
    float z  = 1.f / (1.f + __expf(-(gxz[s] + biz + ghz[s] + bhz)));
    float nn = tanhf(gxn[s] + bin + r * (ghn[s] + bhn));
    snew[s * DDIM + t] = (1.f - z) * nn + z * sp[s * DDIM + t];
  }
  __syncthreads();
  {
    const int s = t >> 4, sub = t & 15;
    float xv[8]; float sm = 0.f, sq = 0.f;
#pragma unroll
    for (int j = 0; j < 8; ++j) { float v = snew[s * DDIM + sub * 8 + j]; xv[j] = v; sm += v; sq += v * v; }
    sm = hred_add16(sm); sq = hred_add16(sq);
    float mu = sm * (1.0f / DDIM);
    float rstd = rsqrtf(sq * (1.0f / DDIM) - mu * mu + LN_EPS);
#pragma unroll
    for (int j = 0; j < 8; ++j) {
      int c = sub * 8 + j;
      nrm[s * DDIM + c] = (xv[j] - mu) * rstd * ln_mlp_w[c] + ln_mlp_b[c];
    }
  }
  __syncthreads();
  float h0[8], h1[8];
#pragma unroll
  for (int s = 0; s < 8; ++s) { h0[s] = 0.f; h1[s] = 0.f; }
  for (int c = 0; c < DDIM; ++c) {
    float wa = mlp_w1[(size_t)c * 2 * DDIM + t];
    float wb = mlp_w1[(size_t)c * 2 * DDIM + t + DDIM];
#pragma unroll
    for (int s = 0; s < 8; ++s) {
      float v = nrm[s * DDIM + c];
      h0[s] += v * wa; h1[s] += v * wb;
    }
  }
  float b1a = mlp_b1[t], b1b = mlp_b1[t + DDIM];
#pragma unroll
  for (int s = 0; s < 8; ++s) {
    hbuf[s * 2 * DDIM + t]        = fmaxf(h0[s] + b1a, 0.f);
    hbuf[s * 2 * DDIM + t + DDIM] = fmaxf(h1[s] + b1b, 0.f);
  }
  __syncthreads();
  float acc2[8];
#pragma unroll
  for (int s = 0; s < 8; ++s) acc2[s] = 0.f;
  for (int c = 0; c < 2 * DDIM; ++c) {
    float w = mlp_w2[(size_t)c * DDIM + t];
#pragma unroll
    for (int s = 0; s < 8; ++s) acc2[s] += hbuf[s * 2 * DDIM + c] * w;
  }
  float b2 = mlp_b2[t];
#pragma unroll
  for (int s = 0; s < 8; ++s)
    slots[((size_t)b * NSLOT + s) * DDIM + t] = snew[s * DDIM + t] + acc2[s] + b2;
}

extern "C" void kernel_launch(void* const* d_in, const int* in_sizes, int n_in,
                              void* d_out, int out_size, void* d_ws, size_t ws_size,
                              hipStream_t stream) {
  const float* inputs    = (const float*)d_in[0];
  const float* slot_init = (const float*)d_in[1];
  const float* Wq        = (const float*)d_in[2];
  const float* Wk        = (const float*)d_in[3];
  const float* Wv        = (const float*)d_in[4];
  const float* gru_wi    = (const float*)d_in[5];
  const float* gru_wh    = (const float*)d_in[6];
  const float* gru_bi    = (const float*)d_in[7];
  const float* gru_bh    = (const float*)d_in[8];
  const float* ln_in_w   = (const float*)d_in[9];
  const float* ln_in_b   = (const float*)d_in[10];
  const float* ln_sl_w   = (const float*)d_in[11];
  const float* ln_sl_b   = (const float*)d_in[12];
  const float* ln_mlp_w  = (const float*)d_in[13];
  const float* ln_mlp_b  = (const float*)d_in[14];
  const float* mlp_w1    = (const float*)d_in[15];
  const float* mlp_b1    = (const float*)d_in[16];
  const float* mlp_w2    = (const float*)d_in[17];
  const float* mlp_b2    = (const float*)d_in[18];
  float* slots = (float*)d_out;

  // workspace carve-up (~137 MB)
  bf16* kb    = (bf16*)d_ws;
  bf16* vT    = kb + (size_t)BATCH * SEQ * DDIM;
  bf16* attnT = vT + (size_t)BATCH * SEQ * DDIM;
  bf16* qb    = attnT + (size_t)BATCH * KPAD * SEQ;
  bf16* wkp   = qb + (size_t)BATCH * NSLOT * DDIM;
  bf16* wvp   = wkp + FDIM * DDIM;
  float* colsum  = (float*)(wvp + FDIM * DDIM);
  float* updates = colsum + BATCH * KPAD;

  k_permute_weights<<<(FDIM * DDIM + 255) / 256, 256, 0, stream>>>(Wk, Wv, wkp, wvp);
  k_init_slots<<<(BATCH * NSLOT * DDIM + 255) / 256, 256, 0, stream>>>(slot_init, slots);
  k_ln_kv<<<(BATCH * SEQ) / 16, 256, 0, stream>>>(inputs, ln_in_w, ln_in_b, wkp, wvp, kb, vT);
  for (int it = 0; it < 3; ++it) {
    k_qproj<<<BATCH, 128, 0, stream>>>(slots, ln_sl_w, ln_sl_b, Wq, qb, colsum, updates);
    dim3 gB(SEQ / 128, BATCH);
    k_attn<<<gB, 256, 0, stream>>>(kb, qb, attnT, colsum);
    dim3 gC(NSEG, BATCH);
    k_updates<<<gC, 256, 0, stream>>>(attnT, vT, updates);
    k_gru_mlp<<<BATCH, 128, 0, stream>>>(updates, colsum, gru_wi, gru_wh, gru_bi, gru_bh,
                                         ln_mlp_w, ln_mlp_b, mlp_w1, mlp_b1, mlp_w2,
                                         mlp_b2, slots);
  }
}
